// BitLinear_28939489640536
// MI455X (gfx1250) — matrix operand order repaired and checked by reading, not compiled, not run on hardware
//
#include <hip/hip_runtime.h>

typedef _Float16 half8  __attribute__((ext_vector_type(8)));
typedef _Float16 half16 __attribute__((ext_vector_type(16)));
typedef float    float8 __attribute__((ext_vector_type(8)));
typedef int      v4i    __attribute__((vector_size(16)));   // matches builtin's param type

#define BM 128
#define BN 128
#define BK 64
#define LDSS 72   // LDS row stride in halves: 64 + 8 pad (144B, 16B-aligned, bank-spread)

#if defined(__AMDGCN__) && __has_builtin(__builtin_amdgcn_global_load_async_to_lds_b128)
#define HAVE_ASYNC_LDS 1
#else
#define HAVE_ASYNC_LDS 0
#endif

// decode one 2-bit ternary code -> fp16 bits (+1 -> 0x3C00, -1 -> 0xBC00, else 0)
__device__ __forceinline__ unsigned int tern2h(unsigned int c) {
    return (c == 1u) ? 0x3C00u : ((c == 2u) ? 0xBC00u : 0u);
}

// 16B global -> LDS copy, async (ASYNCcnt) when available, else via VGPRs
__device__ __forceinline__ void copy16(const _Float16* __restrict__ g, _Float16* l) {
#if HAVE_ASYNC_LDS
    __builtin_amdgcn_global_load_async_to_lds_b128(
        (__attribute__((address_space(1))) v4i*)(g),
        (__attribute__((address_space(3))) v4i*)(l), 0, 0);
#else
    *(half8*)l = *(const half8*)g;
#endif
}

__device__ __forceinline__ void copy_wait() {
#if HAVE_ASYNC_LDS
#if __has_builtin(__builtin_amdgcn_s_wait_asynccnt)
    __builtin_amdgcn_s_wait_asynccnt(0);
#else
    asm volatile("s_wait_asynccnt 0" ::: "memory");
#endif
#endif
}

// ---------------- one-shot ternary decode: packed int32 -> fp16 W in scratch --------
__global__ __launch_bounds__(256)
void decode_w_kernel(const int* __restrict__ packed, _Float16* __restrict__ wdec, int words) {
    int t = blockIdx.x * 256 + threadIdx.x;
    if (t >= words) return;
    unsigned int w = (unsigned int)packed[t];
    unsigned int u[8];
    #pragma unroll
    for (int q = 0; q < 8; ++q) {
        unsigned int clo = (w >> (q * 4))     & 3u;
        unsigned int chi = (w >> (q * 4 + 2)) & 3u;
        u[q] = tern2h(clo) | (tern2h(chi) << 16);
    }
    uint4* d = (uint4*)(wdec + (size_t)t * 16);   // word t covers flat halves [16t,16t+16)
    d[0] = make_uint4(u[0], u[1], u[2], u[3]);
    d[1] = make_uint4(u[4], u[5], u[6], u[7]);
}

// ---------------- main WMMA GEMM: C[T,N] = X[T,K] * W[N,K]^T + bias ------------------
// Double-buffered LDS tiles; async global->LDS fills overlap the WMMA stream.
template <bool PRE>   // PRE: W already decoded to fp16 (wfp16); else decode in-kernel
__global__ __launch_bounds__(256)
void bitlinear_wmma_f16(const _Float16* __restrict__ x,
                        const int*      __restrict__ packed,
                        const _Float16* __restrict__ wfp16,
                        const _Float16* __restrict__ bias,
                        _Float16*       __restrict__ out,
                        int T, int K, int N)
{
    __shared__ _Float16 sx[2][BM * LDSS];   // X tiles, (m, k) row-major
    __shared__ _Float16 sw[2][BN * LDSS];   // W tiles, (n, k) row-major == B column data

    const int tid  = threadIdx.x;
    const int lane = tid & 31;
    const int wave = tid >> 5;        // 0..7
    const int wm   = wave >> 2;       // 0..1 : 64-row M slab
    const int wn   = wave & 3;        // 0..3 : 32-col N slab

    const int m0 = blockIdx.y * BM;
    const int n0 = blockIdx.x * BN;
    const int kwords = K >> 4;

    float8 acc[4][2];
    #pragma unroll
    for (int i = 0; i < 4; ++i)
        #pragma unroll
        for (int j = 0; j < 2; ++j)
            acc[i][j] = float8{};

    const int fr = lane & 15;            // fragment row (M for A) / col (N for B)
    const int kA = (lane >> 4) << 3;     // A: +0 or +8 K-offset group
    const int kB = (lane >> 4) << 3;     // B: +0 or +8 K-offset group

    // ---- branchless tile fills (rows clamped; OOB rows produce unused garbage) ----
    auto fill_x = [&](int kt, _Float16* dst) {
        #pragma unroll
        for (int p = 0; p < 4; ++p) {
            int c   = tid + p * 256;
            int row = c >> 3;
            int cc  = (c & 7) << 3;
            int gr  = m0 + row;
            gr = gr < T ? gr : (T - 1);
            copy16(x + (size_t)gr * K + kt + cc, dst + row * LDSS + cc);
        }
    };
    auto fill_w = [&](int kt, _Float16* dst) {
        if (PRE) {
            #pragma unroll
            for (int p = 0; p < 4; ++p) {
                int c   = tid + p * 256;
                int row = c >> 3;
                int cc  = (c & 7) << 3;
                int gn  = n0 + row;
                gn = gn < N ? gn : (N - 1);
                copy16(wfp16 + (size_t)gn * K + kt + cc, dst + row * LDSS + cc);
            }
        } else {
            int row  = tid >> 1;
            int hsel = tid & 1;
            int gn   = n0 + row;
            gn = gn < N ? gn : (N - 1);
            int wb   = gn * kwords + (kt >> 4) + hsel * 2;
            unsigned int w0 = (unsigned int)packed[wb];
            unsigned int w1 = (unsigned int)packed[wb + 1];
            unsigned int* d = (unsigned int*)(dst + row * LDSS) + hsel * 16;
            #pragma unroll
            for (int wsel = 0; wsel < 2; ++wsel) {
                unsigned int w = wsel ? w1 : w0;
                unsigned int u[8];
                #pragma unroll
                for (int q = 0; q < 8; ++q) {
                    unsigned int clo = (w >> (q * 4))     & 3u;
                    unsigned int chi = (w >> (q * 4 + 2)) & 3u;
                    u[q] = tern2h(clo) | (tern2h(chi) << 16);
                }
                *(uint4*)(d + wsel * 8)     = make_uint4(u[0], u[1], u[2], u[3]);
                *(uint4*)(d + wsel * 8 + 4) = make_uint4(u[4], u[5], u[6], u[7]);
            }
        }
    };

    // ---- prologue: fill buffer 0 ----
    fill_x(0, sx[0]);
    fill_w(0, sw[0]);
    copy_wait();
    __syncthreads();

    int cur = 0;
    for (int kt = 0; kt < K; kt += BK) {
        // ---- kick off next tile's async fills into the other buffer ----
        int nxt = kt + BK;
        if (nxt < K) {
            fill_x(nxt, sx[cur ^ 1]);
            fill_w(nxt, sw[cur ^ 1]);
        }

        // ---- compute on current buffer: 2 K-steps of 32, 8 WMMA each ----
        const _Float16* bx = sx[cur];
        const _Float16* bw = sw[cur];
        #pragma unroll
        for (int ks = 0; ks < BK; ks += 32) {
            half16 af[4];
            #pragma unroll
            for (int i = 0; i < 4; ++i) {
                const _Float16* base = bx + (wm * 64 + i * 16 + fr) * LDSS + ks + kA;
                half8 lo = *(const half8*)(base);         // K + 0..7  (lane group)
                half8 hi = *(const half8*)(base + 16);    // K + 16..23
                af[i] = __builtin_shufflevector(lo, hi,
                        0,1,2,3,4,5,6,7,8,9,10,11,12,13,14,15);
            }
            half16 bf[2];
            #pragma unroll
            for (int j = 0; j < 2; ++j) {
                const _Float16* base = bw + (wn * 32 + j * 16 + fr) * LDSS + ks + kB;
                half8 lo = *(const half8*)(base);         // K + 0..7  (lane group)
                half8 hi = *(const half8*)(base + 16);    // K + 16..23
                bf[j] = __builtin_shufflevector(lo, hi,
                        0,1,2,3,4,5,6,7,8,9,10,11,12,13,14,15);
            }
            #pragma unroll
            for (int i = 0; i < 4; ++i)
                #pragma unroll
                for (int j = 0; j < 2; ++j)
                    acc[i][j] = __builtin_amdgcn_wmma_f32_16x16x32_f16(
                        false, af[i], false, bf[j], (short)0, acc[i][j], false, false);
        }

        copy_wait();        // next-buffer async fills complete
        __syncthreads();    // all waves done reading cur / fills visible
        cur ^= 1;
    }

    // ---- epilogue: C layout lane=N, vgpr r -> M = 8*(lane>=16)+r ----
    const int colbase = n0 + wn * 32;
    const int rsel    = (lane >> 4) << 3;
    #pragma unroll
    for (int j = 0; j < 2; ++j) {
        int col = colbase + j * 16 + fr;
        if (col < N) {
            float bj = (float)bias[col];
            #pragma unroll
            for (int i = 0; i < 4; ++i) {
                int rbase = m0 + wm * 64 + i * 16 + rsel;
                #pragma unroll
                for (int r = 0; r < 8; ++r) {
                    int row = rbase + r;
                    if (row < T)
                        out[(size_t)row * N + col] = (_Float16)(acc[i][j][r] + bj);
                }
            }
        }
    }
}

extern "C" void kernel_launch(void* const* d_in, const int* in_sizes, int n_in,
                              void* d_out, int out_size, void* d_ws, size_t ws_size,
                              hipStream_t stream) {
    const _Float16* x      = (const _Float16*)d_in[0];
    const int*      packed = (const int*)d_in[1];
    const _Float16* bias   = (const _Float16*)d_in[2];
    _Float16*       out    = (_Float16*)d_out;

    const int N = in_sizes[2];                                  // OUT features
    const int K = (int)(((long long)in_sizes[1] * 16) / N);     // IN features
    const int T = in_sizes[0] / K;                              // tokens

    dim3 grid((N + BN - 1) / BN, (T + BM - 1) / BM);
    dim3 block(256);

    const size_t wdec_bytes = (size_t)N * (size_t)K * sizeof(_Float16);
    if (ws_size >= wdec_bytes) {
        _Float16* wdec = (_Float16*)d_ws;
        int words = N * (K >> 4);
        decode_w_kernel<<<(words + 255) / 256, 256, 0, stream>>>(packed, wdec, words);
        bitlinear_wmma_f16<true><<<grid, block, 0, stream>>>(x, packed, wdec, bias, out, T, K, N);
    } else {
        bitlinear_wmma_f16<false><<<grid, block, 0, stream>>>(x, packed, nullptr, bias, out, T, K, N);
    }
}
